// MRConv4d_65180423685328
// MI455X (gfx1250) — matrix operand order repaired and checked by reading, not compiled, not run on hardware
//
#include <hip/hip_runtime.h>
#include <hip/hip_bf16.h>

// MRConv4d fused kernel for gfx1250 (MI455X):
//   x_j = relu(max over 4 circular-shifted neighbors (x_shift - x))
//   y   = GELU( BN( W @ [x; x_j] + b ) )
// GEMM: M=192 (Cout), K=384 (2C), N=131072 (B*H*W), done with
// v_wmma_f32_16x16x32_bf16 using a bf16 hi/lo split for ~f32 accuracy.
// Weight tiles are staged into LDS with GLOBAL_LOAD_ASYNC_TO_LDS_B128
// (ASYNCcnt) when the toolchain exposes the builtins.

typedef __attribute__((ext_vector_type(16))) __bf16 v16bf;
typedef __attribute__((ext_vector_type(8)))  float  v8f;
typedef __attribute__((ext_vector_type(4)))  int    v4i;

// Address-space-qualified views for the async-LDS builtin
// (param 0 is v4i in addrspace(1), param 1 is v4i in addrspace(3)).
typedef __attribute__((address_space(1))) v4i gas_v4i;
typedef __attribute__((address_space(3))) v4i las_v4i;

#define Bsz   32
#define Cin   192
#define Hh    64
#define Ww    64
#define HW    4096
#define COUT  192
#define K2    384
#define KSH   2
#define BN_TILE 128

#if __has_builtin(__builtin_amdgcn_global_load_async_to_lds_b128) && \
    __has_builtin(__builtin_amdgcn_s_wait_asynccnt)
#define USE_ASYNC_LDS 1
#else
#define USE_ASYNC_LDS 0
#endif

union Frag {
  v16bf v;
  uint4 q[2];
};

__device__ __forceinline__ unsigned short f32_to_bf16(float f) {
  unsigned u = __float_as_uint(f);
  unsigned r = 0x7FFFu + ((u >> 16) & 1u);   // round-to-nearest-even
  return (unsigned short)((u + r) >> 16);
}
__device__ __forceinline__ float bf16_to_f32(unsigned short h) {
  return __uint_as_float(((unsigned)h) << 16);
}

// 16-byte global -> LDS copy; async DMA (ASYNCcnt) when available.
__device__ __forceinline__ void copy16_g2l(const void* g, void* l) {
#if USE_ASYNC_LDS
  __builtin_amdgcn_global_load_async_to_lds_b128(
      (gas_v4i*)(void*)const_cast<void*>(g),
      (las_v4i*)l, 0, 0);
#else
  *(uint4*)l = *(const uint4*)g;
#endif
}

__device__ __forceinline__ void wait_async_copies() {
#if USE_ASYNC_LDS
  __builtin_amdgcn_s_wait_asynccnt(0);
#endif
}

// ---------------------------------------------------------------------------
// Prep: split conv_w into bf16 hi/lo pair; fold conv_b + BN into alpha/beta0.
// ---------------------------------------------------------------------------
__global__ void prep_kernel(const float* __restrict__ w,
                            const float* __restrict__ conv_b,
                            const float* __restrict__ gamma,
                            const float* __restrict__ beta,
                            const float* __restrict__ mean,
                            const float* __restrict__ var,
                            unsigned short* __restrict__ whi,
                            unsigned short* __restrict__ wlo,
                            float* __restrict__ alpha,
                            float* __restrict__ beta0) {
  int i = blockIdx.x * blockDim.x + threadIdx.x;
  if (i < COUT * K2) {
    float f = w[i];
    unsigned short h = f32_to_bf16(f);
    whi[i] = h;
    wlo[i] = f32_to_bf16(f - bf16_to_f32(h));
  }
  if (i < COUT) {
    float inv = gamma[i] * rsqrtf(var[i] + 1e-5f);
    alpha[i] = inv;
    beta0[i] = conv_b[i] * inv + beta[i] - mean[i] * inv;
  }
}

// ---------------------------------------------------------------------------
// Main fused GEMM. Grid: (HW/128, B). Block: 256 threads = 8 wave32.
// Wave layout 4x2: wave handles 48 (M) x 64 (N) = 12 tiles of 16x16.
// ---------------------------------------------------------------------------
__global__ void __launch_bounds__(256)
mrconv_kernel(const float* __restrict__ xg,
              const unsigned short* __restrict__ whi,
              const unsigned short* __restrict__ wlo,
              const float* __restrict__ alpha,
              const float* __restrict__ beta0,
              float* __restrict__ out) {
  __shared__ __align__(16) unsigned short lw_hi[COUT * 32];     // [m][k]
  __shared__ __align__(16) unsigned short lw_lo[COUT * 32];
  __shared__ __align__(16) unsigned short lx_hi[BN_TILE * 32];  // [n][k]
  __shared__ __align__(16) unsigned short lx_lo[BN_TILE * 32];

  const int tid   = threadIdx.x;
  const int bq    = blockIdx.y;
  const int ntile = blockIdx.x * BN_TILE;

  const int wave   = tid >> 5;
  const int lane   = tid & 31;
  const int wm     = wave & 3;              // M block of 48 rows
  const int wn     = wave >> 2;             // N block of 64 cols
  const int lrow   = lane & 15;
  const int khalfA = (lane < 16) ? 0 : 8;   // A: lanes16-31 start at K=8
  const int khalfB = (lane < 16) ? 0 : 8;   // B: lanes16-31 start at K=8

  v8f zero = {};
  v8f acc[3][4];
#pragma unroll
  for (int mt = 0; mt < 3; ++mt)
#pragma unroll
    for (int nt = 0; nt < 4; ++nt)
      acc[mt][nt] = zero;

  for (int kk = 0; kk < 12; ++kk) {
    // ---- Issue weight-tile DMA first: 192x32 bf16 (hi+lo) = 24 KB.
    // 768 16-byte chunks per buffer; 3 per thread per buffer.
    const int kbase = kk * 32;            // K offset in halves
#pragma unroll
    for (int i = 0; i < 3; ++i) {
      int idx  = i * 256 + tid;           // 0..767
      int m    = idx >> 2;                // weight row
      int part = (idx & 3) << 3;          // 0,8,16,24 halves within row slice
      copy16_g2l(&whi[m * K2 + kbase + part], &lw_hi[(m << 5) + part]);
      copy16_g2l(&wlo[m * K2 + kbase + part], &lw_lo[(m << 5) + part]);
    }

    // ---- Stage X tile 32x128 (hi+lo), transposed to [n][k] in LDS.
    // This VALU/LDS work overlaps the in-flight async weight DMA.
    const int  cbase    = (kk % 6) * 32;
    const bool xj_phase = (kk >= 6);
#pragma unroll
    for (int i = 0; i < 16; ++i) {
      int idx = i * 256 + tid;            // 0..4095
      int cl  = idx >> 7;                 // 0..31  (k within step)
      int n   = idx & 127;                // 0..127 (column)
      const float* row = xg + ((size_t)(bq * Cin + cbase + cl)) * HW;
      int p = ntile + n;
      float v;
      if (!xj_phase) {
        v = row[p];
      } else {
        int h = p >> 6, w = p & 63;
        float ctr = row[p];
        float up  = row[(((h + Hh - KSH) & 63) << 6) + w];
        float dn  = row[(((h + KSH) & 63) << 6) + w];
        float lf  = row[(h << 6) + ((w + Ww - KSH) & 63)];
        float rt  = row[(h << 6) + ((w + KSH) & 63)];
        v = fmaxf(fmaxf(up, dn), fmaxf(lf, rt)) - ctr;
        v = fmaxf(v, 0.0f);
      }
      unsigned short hh = f32_to_bf16(v);
      lx_hi[(n << 5) + cl] = hh;
      lx_lo[(n << 5) + cl] = f32_to_bf16(v - bf16_to_f32(hh));
    }

    wait_async_copies();                  // s_wait_asynccnt 0
    __syncthreads();

    // ---- B fragments (4 N-tiles), K in table order (same as A) ----
    Frag bh[4], bl[4];
#pragma unroll
    for (int nt = 0; nt < 4; ++nt) {
      int ncol = wn * 64 + nt * 16 + lrow;
      const uint4* ph = (const uint4*)&lx_hi[(ncol << 5) + khalfB];
      const uint4* pl = (const uint4*)&lx_lo[(ncol << 5) + khalfB];
      bh[nt].q[0] = ph[0]; bh[nt].q[1] = ph[2];
      bl[nt].q[0] = pl[0]; bl[nt].q[1] = pl[2];
    }

    // ---- A fragments + WMMA: D += Wh*Xh + Wh*Xl + Wl*Xh ----
#pragma unroll
    for (int mt = 0; mt < 3; ++mt) {
      int mrow = wm * 48 + mt * 16 + lrow;
      Frag ah, al;
      ah.q[0] = *(const uint4*)&lw_hi[(mrow << 5) + khalfA];
      ah.q[1] = *(const uint4*)&lw_hi[(mrow << 5) + 16 + khalfA];
      al.q[0] = *(const uint4*)&lw_lo[(mrow << 5) + khalfA];
      al.q[1] = *(const uint4*)&lw_lo[(mrow << 5) + 16 + khalfA];
#pragma unroll
      for (int nt = 0; nt < 4; ++nt) {
        acc[mt][nt] = __builtin_amdgcn_wmma_f32_16x16x32_bf16(
            false, ah.v, false, bh[nt].v, (short)0, acc[mt][nt], false, false);
        acc[mt][nt] = __builtin_amdgcn_wmma_f32_16x16x32_bf16(
            false, ah.v, false, bl[nt].v, (short)0, acc[mt][nt], false, false);
        acc[mt][nt] = __builtin_amdgcn_wmma_f32_16x16x32_bf16(
            false, al.v, false, bh[nt].v, (short)0, acc[mt][nt], false, false);
      }
    }
    __syncthreads();
  }

  // ---- Epilogue: folded bias+BN affine, then exact GELU ----
  const int mhalf = (lane < 16) ? 0 : 8;   // D layout: M = r + 8*(lane>=16)
#pragma unroll
  for (int mt = 0; mt < 3; ++mt) {
#pragma unroll
    for (int nt = 0; nt < 4; ++nt) {
      int nglob = ntile + wn * 64 + nt * 16 + lrow;
#pragma unroll
      for (int r = 0; r < 8; ++r) {
        int m = wm * 48 + mt * 16 + mhalf + r;
        float y = acc[mt][nt][r] * alpha[m] + beta0[m];
        float g = 0.5f * y * (1.0f + erff(y * 0.70710678118654752f));
        out[((size_t)(bq * COUT + m)) * HW + nglob] = g;
      }
    }
  }
}

// ---------------------------------------------------------------------------
extern "C" void kernel_launch(void* const* d_in, const int* in_sizes, int n_in,
                              void* d_out, int out_size, void* d_ws, size_t ws_size,
                              hipStream_t stream) {
  const float* x      = (const float*)d_in[0];
  const float* conv_w = (const float*)d_in[1];
  const float* conv_b = (const float*)d_in[2];
  const float* gamma  = (const float*)d_in[3];
  const float* beta   = (const float*)d_in[4];
  const float* mean   = (const float*)d_in[5];
  const float* var    = (const float*)d_in[6];

  char* ws = (char*)d_ws;
  unsigned short* whi   = (unsigned short*)ws;                       // 147456 B
  unsigned short* wlo   = (unsigned short*)(ws + COUT * K2 * 2);     // 147456 B
  float*          alpha = (float*)(ws + COUT * K2 * 4);              //    768 B
  float*          beta0 = (float*)(ws + COUT * K2 * 4 + COUT * 4);   //    768 B

  prep_kernel<<<(COUT * K2 + 255) / 256, 256, 0, stream>>>(
      conv_w, conv_b, gamma, beta, mean, var, whi, wlo, alpha, beta0);

  dim3 grid(HW / BN_TILE, Bsz);   // 32 x 32 workgroups
  mrconv_kernel<<<grid, 256, 0, stream>>>(x, whi, wlo, alpha, beta0,
                                          (float*)d_out);
}
